// SOC_Mamba_V1_22230750724314
// MI455X (gfx1250) — matrix operand order repaired and checked by reading, not compiled, not run on hardware
//
#include <hip/hip_runtime.h>
#include <hip/hip_bf16.h>

// ---------------- CDNA5 WMMA types ----------------
typedef __attribute__((ext_vector_type(16))) _Float16 v16h;
typedef __attribute__((ext_vector_type(8)))  _Float16 v8h;
typedef __attribute__((ext_vector_type(8)))  float    v8f;

union H16 { v16h v; v8h h[2]; };

// Model constants
#define BT     4096      // B*T tokens
#define TT     1024      // T
#define DM     256       // d_model
#define ED     512       // d_inner
#define NST    16        // d_state
#define DTR    16        // dt_rank
#define DTRP   32        // dt_rank padded to one WMMA K-step
#define DBCW   48        // dt_rank + 2*d_state

// ---------------- fragment loaders (ISA 7.12.2 layouts) ----------------
// A: 16x32 f16 (MxK).  lane<16: m=lane, K = {k0+8hf..+7} U {k0+16+8hf..+7}
__device__ __forceinline__ v16h load_a_frag(const _Float16* A, int lda, int mbase,
                                            int k0, int lane) {
  int r = lane & 15, hf = lane >> 4;
  const _Float16* p = A + (size_t)(mbase + r) * lda + k0 + 8 * hf;
  H16 u;
  u.h[0] = *(const v8h*)p;         // K = k0+8hf .. +7
  u.h[1] = *(const v8h*)(p + 16);  // K = k0+16+8hf .. +7
  return u.v;
}
// B: 32x16 f16 (KxN), sourced from W stored row-major (N x K): B[k][n]=W[n][k].
// lane: n = lane&15 ; K = {k0+8hf..+7} U {k0+16+8hf..+7} (same as A)
__device__ __forceinline__ v16h load_b_frag(const _Float16* W, int ldw, int nbase,
                                            int k0, int lane) {
  int r = lane & 15, hf = lane >> 4;
  const _Float16* p = W + (size_t)(nbase + r) * ldw + k0 + 8 * hf;
  H16 u;
  u.h[0] = *(const v8h*)p;
  u.h[1] = *(const v8h*)(p + 16);
  return u.v;
}

// ---------------- register-blocked WMMA GEMM -------------------------------
//  C[m][n] = sum_k A[m][k]*W[n][k]
//  One wave computes a 16 x (16*NB) strip: A fragment loaded once per K-step,
//  NB back-to-back WMMAs against it (A reuse = NB).
// mode 0: store C
// mode 1: C = softplus(acc + bias[n])        (dt_proj)
// mode 2: C[m][n] += acc                     (out_proj residual into h)
template <int NB>
__global__ void gemm_wmma_k(const _Float16* __restrict__ A,
                            const _Float16* __restrict__ W,
                            float* __restrict__ C,
                            int M, int N, int K, int mode,
                            const float* __restrict__ bias) {
  int wid  = threadIdx.x >> 5;
  int lane = threadIdx.x & 31;
  int tile = blockIdx.x * (blockDim.x >> 5) + wid;
  int ntb  = N / (16 * NB);
  int total = (M >> 4) * ntb;
  if (tile >= total) return;
  int mt  = tile / ntb;
  int nt0 = (tile % ntb) * NB;

  v8f zero = {};
  v8f acc[NB];
#pragma unroll
  for (int q = 0; q < NB; ++q) acc[q] = zero;

  for (int k0 = 0; k0 < K; k0 += 32) {
    v16h a = load_a_frag(A, K, mt * 16, k0, lane);
#pragma unroll
    for (int q = 0; q < NB; ++q) {
      v16h b = load_b_frag(W, K, (nt0 + q) * 16, k0, lane);
      acc[q] = __builtin_amdgcn_wmma_f32_16x16x32_f16(
          /*neg_a=*/false, a, /*neg_b=*/false, b,
          /*c_mod=*/(short)0, acc[q], /*reuse_a=*/false, /*reuse_b=*/false);
    }
  }

  int r = lane & 15, hf = lane >> 4;
#pragma unroll
  for (int q = 0; q < NB; ++q) {
    int n = (nt0 + q) * 16 + r;
#pragma unroll
    for (int j = 0; j < 8; ++j) {
      int m = mt * 16 + j + 8 * hf;          // C/D layout: VGPR j -> row j + 8*hf
      size_t idx = (size_t)m * N + n;
      float v = acc[q][j];
      if (mode == 1) {
        v += bias[n];
        v = (v > 20.f) ? v : log1pf(__expf(v));   // softplus
        C[idx] = v;
      } else if (mode == 2) {
        C[idx] += v;                              // residual accumulate
      } else {
        C[idx] = v;
      }
    }
  }
}

// ---------------- weight conversion ----------------
__global__ void cvt_f16_k(const float* __restrict__ s, _Float16* __restrict__ d, int n) {
  int i = blockIdx.x * blockDim.x + threadIdx.x;
  if (i < n) d[i] = (_Float16)s[i];
}
// dt_proj (512x16) -> f16 (512x32) zero-padded K
__global__ void pad_dtw_k(const float* __restrict__ s, _Float16* __restrict__ d) {
  int i = blockIdx.x * blockDim.x + threadIdx.x;   // over 512*32
  if (i >= ED * DTRP) return;
  int r = i >> 5, c = i & 31;
  d[i] = (c < DTR) ? (_Float16)s[r * DTR + c] : (_Float16)0.f;
}
// delta_in from dbc (4096x48) -> f16 (4096x32) zero-padded
__global__ void pad_delta_k(const float* __restrict__ dbc, _Float16* __restrict__ d) {
  int i = blockIdx.x * blockDim.x + threadIdx.x;   // over 4096*32
  if (i >= BT * DTRP) return;
  int m = i >> 5, c = i & 31;
  d[i] = (c < DTR) ? (_Float16)dbc[(size_t)m * DBCW + c] : (_Float16)0.f;
}

// ---------------- block reduce (256 threads) ----------------
__device__ __forceinline__ float block_reduce_256(float v, float* red, int j) {
  __syncthreads();
  red[j] = v;
  __syncthreads();
  for (int s = 128; s > 0; s >>= 1) {
    if (j < s) red[j] += red[j + s];
    __syncthreads();
  }
  return red[0];
}

// ---------------- embed: iu-normalize, input_proj(12->256), LayerNorm ------
__global__ void embed_ln_k(const float* __restrict__ x,
                           const float* __restrict__ ium, const float* __restrict__ ius,
                           const float* __restrict__ Wp,  const float* __restrict__ bp,
                           const float* __restrict__ lnw, const float* __restrict__ lnb,
                           float* __restrict__ h) {
  int m = blockIdx.x, j = threadIdx.x;         // token, out-channel
  __shared__ float f[12];
  __shared__ float red[256];
  if (j < 12) {
    float v = x[(size_t)m * 12 + j];
    if (j < 2) v = (v - ium[j]) / (ius[j] + 1e-6f);
    f[j] = v;
  }
  __syncthreads();
  float a = bp[j];
#pragma unroll
  for (int k = 0; k < 12; ++k) a += Wp[j * 12 + k] * f[k];
  float mean = block_reduce_256(a, red, j) * (1.f / DM);
  float d = a - mean;
  float var = block_reduce_256(d * d, red, j) * (1.f / DM);
  h[(size_t)m * DM + j] = d * rsqrtf(var + 1e-5f) * lnw[j] + lnb[j];
}

// ---------------- RMSNorm -> f16 -------------------------------------------
__global__ void rmsnorm16_k(const float* __restrict__ h, const float* __restrict__ w,
                            _Float16* __restrict__ out) {
  int m = blockIdx.x, j = threadIdx.x;
  __shared__ float red[256];
  float v = h[(size_t)m * DM + j];
  float ss = block_reduce_256(v * v, red, j) * (1.f / DM);
  out[(size_t)m * DM + j] = (_Float16)(v * rsqrtf(ss + 1e-5f) * w[j]);
}

// ---------------- causal depthwise conv (k=4) + SiLU -----------------------
// input xb = xz[:, 0:512] (stride 1024); outputs f32 + f16
__global__ void conv_silu_k(const float* __restrict__ xz,
                            const float* __restrict__ cw, const float* __restrict__ cb,
                            float* __restrict__ xo, _Float16* __restrict__ xo16) {
  size_t i = (size_t)blockIdx.x * blockDim.x + threadIdx.x;   // over BT*ED
  if (i >= (size_t)BT * ED) return;
  int e = (int)(i & (ED - 1));
  size_t m = i >> 9;
  int t = (int)(m & (TT - 1));
  size_t b = m >> 10;
  float acc = cb[e];
#pragma unroll
  for (int j = 0; j < 4; ++j) {
    int tt = t - 3 + j;
    if (tt >= 0) acc += cw[e * 4 + j] * xz[((b << 10) + tt) * 1024 + e];
  }
  float s = acc / (1.f + __expf(-acc));     // silu
  xo[i] = s;
  xo16[i] = (_Float16)s;
}

// ---------------- selective scan + D-skip + SiLU(z) gate -------------------
// half-wave (16 lanes) per (b,e); lane&15 = state n; y reduced via shfl_xor.
__global__ void scan_gate_k(const float* __restrict__ delta,
                            const float* __restrict__ xbc,
                            const float* __restrict__ xz,
                            const float* __restrict__ dbc,
                            const float* __restrict__ A_log,
                            const float* __restrict__ Dp,
                            _Float16* __restrict__ yg16) {
  int wid  = threadIdx.x >> 5;
  int lane = threadIdx.x & 31;
  int w = blockIdx.x * (blockDim.x >> 5) + wid;      // 1024 waves total
  int n = lane & 15;
  int b = w >> 8;                                    // 256 channel-pairs per batch
  int e = ((w & 255) << 1) + (lane >> 4);
  float a = -__expf(A_log[e * NST + n]);
  float dskip = Dp[e];
  float h = 0.f;
  for (int t = 0; t < TT; ++t) {
    size_t m = (size_t)b * TT + t;
    float dl = delta[m * ED + e];
    float xv = xbc[m * ED + e];
    float Bn = dbc[m * DBCW + DTR + n];
    float Cn = dbc[m * DBCW + DTR + NST + n];
    h = __expf(dl * a) * h + dl * Bn * xv;
    float p = h * Cn;
#pragma unroll
    for (int o = 1; o < 16; o <<= 1) p += __shfl_xor(p, o, 16);
    if (n == 0) {
      float y = p + dskip * xv;
      float z = xz[m * 1024 + ED + e];
      y *= z / (1.f + __expf(-z));                   // * silu(z)
      yg16[m * ED + e] = (_Float16)y;
    }
  }
}

// ---------------- final RMSNorm + head (256 -> 1) --------------------------
__global__ void head_k(const float* __restrict__ h, const float* __restrict__ w,
                       const float* __restrict__ hw, const float* __restrict__ hb,
                       float* __restrict__ out) {
  int m = blockIdx.x, j = threadIdx.x;
  __shared__ float red[256];
  float v = h[(size_t)m * DM + j];
  float ss = block_reduce_256(v * v, red, j) * (1.f / DM);
  float c = v * rsqrtf(ss + 1e-5f) * w[j] * hw[j];
  float tot = block_reduce_256(c, red, j);
  if (j == 0) out[m] = tot + hb[0];
}

// ---------------- host launcher --------------------------------------------
extern "C" void kernel_launch(void* const* d_in, const int* in_sizes, int n_in,
                              void* d_out, int out_size, void* d_ws, size_t ws_size,
                              hipStream_t stream) {
  (void)in_sizes; (void)n_in; (void)out_size; (void)ws_size;
  const float* X    = (const float*)d_in[0];
  const float* ium  = (const float*)d_in[1];
  const float* ius  = (const float*)d_in[2];
  const float* ipw  = (const float*)d_in[3];
  const float* ipb  = (const float*)d_in[4];
  const float* lnw  = (const float*)d_in[5];
  const float* lnb  = (const float*)d_in[6];
  const float* fnw  = (const float*)d_in[47];
  const float* hw   = (const float*)d_in[48];
  const float* hb   = (const float*)d_in[49];

  char* ws = (char*)d_ws;
  size_t off = 0;
  auto nxt = [&](size_t bytes) { size_t r = off; off += (bytes + 255) & ~(size_t)255; return r; };

  float*    h      = (float*)   (ws + nxt((size_t)BT * DM * 4));
  _Float16* xn16   = (_Float16*)(ws + nxt((size_t)BT * DM * 2));
  float*    xz     = (float*)   (ws + nxt((size_t)BT * 1024 * 4));
  float*    xbc    = (float*)   (ws + nxt((size_t)BT * ED * 4));
  _Float16* xbc16  = (_Float16*)(ws + nxt((size_t)BT * ED * 2));
  float*    dbc    = (float*)   (ws + nxt((size_t)BT * DBCW * 4));
  _Float16* dpad16 = (_Float16*)(ws + nxt((size_t)BT * DTRP * 2));
  float*    delta  = (float*)   (ws + nxt((size_t)BT * ED * 4));
  _Float16* yg16   = (_Float16*)(ws + nxt((size_t)BT * ED * 2));
  _Float16* w_inp[4]; _Float16* w_xp[4]; _Float16* w_dt[4]; _Float16* w_op[4];
  for (int l = 0; l < 4; ++l) {
    w_inp[l] = (_Float16*)(ws + nxt((size_t)1024 * DM * 2));
    w_xp[l]  = (_Float16*)(ws + nxt((size_t)DBCW * ED * 2));
    w_dt[l]  = (_Float16*)(ws + nxt((size_t)ED * DTRP * 2));
    w_op[l]  = (_Float16*)(ws + nxt((size_t)DM * ED * 2));
  }

  // weights -> f16 (recomputed every call: deterministic, capture-safe)
  for (int l = 0; l < 4; ++l) {
    const float* in_proj = (const float*)d_in[7 + 10 * l + 1];
    const float* x_proj  = (const float*)d_in[7 + 10 * l + 4];
    const float* dt_proj = (const float*)d_in[7 + 10 * l + 5];
    const float* out_prj = (const float*)d_in[7 + 10 * l + 9];
    cvt_f16_k<<<(1024 * DM + 255) / 256, 256, 0, stream>>>(in_proj, w_inp[l], 1024 * DM);
    cvt_f16_k<<<(DBCW * ED + 255) / 256, 256, 0, stream>>>(x_proj, w_xp[l], DBCW * ED);
    pad_dtw_k<<<(ED * DTRP + 255) / 256, 256, 0, stream>>>(dt_proj, w_dt[l]);
    cvt_f16_k<<<(DM * ED + 255) / 256, 256, 0, stream>>>(out_prj, w_op[l], DM * ED);
  }

  // embedding + layernorm
  embed_ln_k<<<BT, 256, 0, stream>>>(X, ium, ius, ipw, ipb, lnw, lnb, h);

  for (int l = 0; l < 4; ++l) {
    const float* norm_w  = (const float*)d_in[7 + 10 * l + 0];
    const float* conv_w  = (const float*)d_in[7 + 10 * l + 2];
    const float* conv_b  = (const float*)d_in[7 + 10 * l + 3];
    const float* dt_bias = (const float*)d_in[7 + 10 * l + 6];
    const float* A_log   = (const float*)d_in[7 + 10 * l + 7];
    const float* Dp      = (const float*)d_in[7 + 10 * l + 8];

    // pre-norm -> f16
    rmsnorm16_k<<<BT, 256, 0, stream>>>(h, norm_w, xn16);
    // in_proj: 4096x256 @ (1024x256)^T -> xz 4096x1024   [NB=4: 4096 strip-tiles]
    gemm_wmma_k<4><<<(BT / 16) * (1024 / 64) / 8, 256, 0, stream>>>(
        xn16, w_inp[l], xz, BT, 1024, DM, 0, nullptr);
    // causal depthwise conv + silu on xb half
    conv_silu_k<<<(BT * ED) / 256, 256, 0, stream>>>(xz, conv_w, conv_b, xbc, xbc16);
    // x_proj: 4096x512 @ (48x512)^T -> dbc 4096x48   [NB=1: 768 tiles]
    gemm_wmma_k<1><<<(BT / 16) * (DBCW / 16) / 8, 256, 0, stream>>>(
        xbc16, w_xp[l], dbc, BT, DBCW, ED, 0, nullptr);
    // pad delta_in to K=32 f16
    pad_delta_k<<<(BT * DTRP) / 256, 256, 0, stream>>>(dbc, dpad16);
    // dt_proj + bias + softplus: 4096x32 @ (512x32)^T -> delta 4096x512  [NB=4]
    gemm_wmma_k<4><<<(BT / 16) * (ED / 64) / 8, 256, 0, stream>>>(
        dpad16, w_dt[l], delta, BT, ED, DTRP, 1, dt_bias);
    // selective scan + gate -> yg16 (1024 waves)
    scan_gate_k<<<128, 256, 0, stream>>>(delta, xbc, xz, dbc, A_log, Dp, yg16);
    // out_proj + residual: 4096x512 @ (256x512)^T  accum into h  [NB=4]
    gemm_wmma_k<4><<<(BT / 16) * (DM / 64) / 8, 256, 0, stream>>>(
        yg16, w_op[l], h, BT, DM, ED, 2, nullptr);
  }

  // final RMSNorm + head
  head_k<<<BT, 256, 0, stream>>>(h, fnw, hw, hb, (float*)d_out);
}
